// AttnCorrelation_63007170232539
// MI455X (gfx1250) — matrix operand order repaired and checked by reading, compiled, not run on hardware
//
#include <hip/hip_runtime.h>
#include <hip/hip_bf16.h>
#include <math.h>

typedef __attribute__((ext_vector_type(16))) _Float16 v16h;
typedef __attribute__((ext_vector_type(8)))  _Float16 v8h;
typedef __attribute__((ext_vector_type(8)))  float    v8f;

#define BD    8
#define CD    256
#define HD    96
#define WD    128
#define HWD   (HD * WD)      // 12288
#define CR    32
#define DRAD  4
#define NSH   9              // 2*DRAD+1
#define NSH2  (NSH * NSH)    // 81

// ---------------------------------------------------------------------------
// Kernel 1: 1x1 conv (per-pixel channel GEMM) via WMMA f32 <- f16 x f16,
// with LDS-staged, transposed operands.
//   dst[b][p][o] = sum_c feat[b][c][p] * Wt[o][c] + bias[o],  o in [0,32)
// Block = 128 threads (4 waves), owns 64 pixels. Per K-step (K=32 channels):
//   - coalesced global_load_b128 of the [32ch x 64px] f32 tile,
//   - f32->f16 convert + transpose into LDS [px][ch],
//   - each wave's A fragment = two 16B ds_load_b128,
//   - B fragments from LDS-resident f16 weights (staged once per block),
//   - 2x v_wmma_f32_16x16x32_f16 (N-tiles o=0..15 and o=16..31).
// ---------------------------------------------------------------------------
__global__ __launch_bounds__(128)
void conv1x1_wmma(const float* __restrict__ feat, const float* __restrict__ Wt,
                  const float* __restrict__ bias, float* __restrict__ dst)
{
    __shared__ __align__(16) _Float16 sW[CR * CD];   // [o][c] f16  (16 KB)
    __shared__ __align__(16) _Float16 sA[64 * CR];   // [px][ch] f16 (4 KB)

    const int t    = threadIdx.x;
    const int lane = t & 31;
    const int wid  = t >> 5;
    const int blocksPerBatch = HWD / 64;             // 192
    const int b  = blockIdx.x / blocksPerBatch;
    const int pB = (blockIdx.x % blocksPerBatch) * 64;

    const float* fb = feat + (size_t)b * CD * HWD + pB;

    // ---- Stage weights once: 8192 floats, 64 per thread, f32 -> f16 ----
    {
        const int base = t * 64;
        #pragma unroll
        for (int e = 0; e < 64; e += 4) {
            float4 v = *(const float4*)(Wt + base + e);
            sW[base + e + 0] = (_Float16)v.x;
            sW[base + e + 1] = (_Float16)v.y;
            sW[base + e + 2] = (_Float16)v.z;
            sW[base + e + 3] = (_Float16)v.w;
        }
    }

    const int m    = lane & 15;          // M (pixel) / N (output) index
    const int ksel = lane >> 4;          // lane-half of K
    const int pLoc = wid * 16 + m;       // pixel within the block's 64

    // Staging assignment: each thread owns 16 pixels of one channel.
    const int chS  = t >> 2;             // 0..31
    const int px0  = (t & 3) * 16;       // 0,16,32,48

    v8f acc0 = {};
    v8f acc1 = {};

    for (int kstep = 0; kstep < 8; ++kstep) {
        const int c0 = kstep * 32;
        __syncthreads();   // LDS tile free (also covers the W staging on k=0)

        // ---- Stage A tile: coalesced b128 loads, convert, transpose ----
        {
            const float* src = fb + (size_t)(c0 + chS) * HWD + px0;
            if (kstep < 7)   // prefetch next K-step's channel row
                __builtin_prefetch(src + (size_t)32 * HWD, 0, 1);
            #pragma unroll
            for (int e = 0; e < 16; e += 4) {
                float4 v = *(const float4*)(src + e);
                sA[(px0 + e + 0) * CR + chS] = (_Float16)v.x;
                sA[(px0 + e + 1) * CR + chS] = (_Float16)v.y;
                sA[(px0 + e + 2) * CR + chS] = (_Float16)v.z;
                sA[(px0 + e + 3) * CR + chS] = (_Float16)v.w;
            }
        }
        __syncthreads();

        // ---- A fragment (16-bit A 16x32 layout, contiguous K order):
        //      K = ksel*16 + e
        //      -> two channel-contiguous 8xf16 runs = two ds_load_b128
        v8h alo = *(const v8h*)(sA + pLoc * CR + ksel * 16);
        v8h ahi = *(const v8h*)(sA + pLoc * CR + ksel * 16 + 8);
        v16h a;
        #pragma unroll
        for (int e = 0; e < 8; ++e) { a[e] = alo[e]; a[e + 8] = ahi[e]; }

        // ---- B fragments (32x16 layout): K = ksel*16 + e, contiguous in c
        const _Float16* w0 = sW + (size_t)m        * CD + c0 + ksel * 16;
        const _Float16* w1 = sW + (size_t)(m + 16) * CD + c0 + ksel * 16;
        v8h b0lo = *(const v8h*)(w0);
        v8h b0hi = *(const v8h*)(w0 + 8);
        v8h b1lo = *(const v8h*)(w1);
        v8h b1hi = *(const v8h*)(w1 + 8);
        v16h bf0, bf1;
        #pragma unroll
        for (int e = 0; e < 8; ++e) {
            bf0[e] = b0lo[e]; bf0[e + 8] = b0hi[e];
            bf1[e] = b1lo[e]; bf1[e + 8] = b1hi[e];
        }

        acc0 = __builtin_amdgcn_wmma_f32_16x16x32_f16(
            false, a, false, bf0, (short)0, acc0, false, false);
        acc1 = __builtin_amdgcn_wmma_f32_16x16x32_f16(
            false, a, false, bf1, (short)0, acc1, false, false);
    }

    // Bias + store. C/D layout: VGPR r -> M = r + 8*ksel, N = lane&15.
    const float bv0 = bias[m];
    const float bv1 = bias[m + 16];
    float* drow = dst + (size_t)b * HWD * CR;
    #pragma unroll
    for (int r = 0; r < 8; ++r) {
        const int p = pB + wid * 16 + r + ksel * 8;
        drow[(size_t)p * CR + m]      = acc0[r] + bv0;
        drow[(size_t)p * CR + m + 16] = acc1[r] + bv1;
    }
}

// ---------------------------------------------------------------------------
// Kernel 2: 81-shift correlation + softmax over a size-1 axis.
//   corr = mean_o q[b,o,h,w] * k[b,o,h+i-4,w+j-4]   (zero padded)
//   out  = exp(corr - max)/sum over a singleton axis == exp(c - c)
// k per batch is 1.5 MB -> L2 resident with 81x reuse.
// ---------------------------------------------------------------------------
__global__ __launch_bounds__(256)
void corr_softmax(const float* __restrict__ q, const float* __restrict__ k,
                  float* __restrict__ out)
{
    const int idx = blockIdx.x * 256 + threadIdx.x;   // over B*HW
    const int b = idx / HWD;
    const int p = idx - b * HWD;
    const int h = p / WD;
    const int w = p - h * WD;

    float qr[CR];
    const float* qrow = q + ((size_t)b * HWD + p) * CR;
    #pragma unroll
    for (int e = 0; e < CR; e += 4) {
        float4 v = *(const float4*)(qrow + e);
        qr[e] = v.x; qr[e + 1] = v.y; qr[e + 2] = v.z; qr[e + 3] = v.w;
    }

    const float* kb = k + (size_t)b * HWD * CR;
    for (int i = 0; i < NSH; ++i) {
        const int hh = h + i - DRAD;
        for (int j = 0; j < NSH; ++j) {
            const int ww = w + j - DRAD;
            float c = 0.0f;
            if ((unsigned)hh < (unsigned)HD && (unsigned)ww < (unsigned)WD) {
                const float* krow = kb + (size_t)(hh * WD + ww) * CR;
                #pragma unroll
                for (int e = 0; e < CR; e += 4) {
                    float4 v = *(const float4*)(krow + e);
                    c += qr[e] * v.x + qr[e + 1] * v.y
                       + qr[e + 2] * v.z + qr[e + 3] * v.w;
                }
                c *= (1.0f / CR);
            }
            // softmax along a singleton axis: exp(x - max)/sum == exp(c - c)
            // (== 1.0 for finite c; propagates NaN exactly like the reference)
            out[((size_t)b * NSH2 + (i * NSH + j)) * HWD + p] = __expf(c - c);
        }
    }
}

// Fallback if the workspace is too small: the result is identically 1.0f.
__global__ __launch_bounds__(256)
void fill_ones(float* __restrict__ out, int n)
{
    int i = blockIdx.x * 256 + threadIdx.x;
    if (i < n) out[i] = 1.0f;
}

extern "C" void kernel_launch(void* const* d_in, const int* in_sizes, int n_in,
                              void* d_out, int out_size, void* d_ws, size_t ws_size,
                              hipStream_t stream) {
    (void)in_sizes; (void)n_in;
    const float* feat1 = (const float*)d_in[0];
    const float* feat2 = (const float*)d_in[1];
    const float* Wq    = (const float*)d_in[2];
    const float* bq    = (const float*)d_in[3];
    const float* Wk    = (const float*)d_in[4];
    const float* bk    = (const float*)d_in[5];
    float* out = (float*)d_out;

    const size_t qk_elems = (size_t)BD * HWD * CR;        // 3.1M floats each
    if (ws_size < 2 * qk_elems * sizeof(float)) {
        // Not enough scratch: emit the (numerically identical) constant result.
        fill_ones<<<(out_size + 255) / 256, 256, 0, stream>>>(out, out_size);
        return;
    }
    float* qws = (float*)d_ws;
    float* kws = qws + qk_elems;

    dim3 g1(BD * (HWD / 64));                             // 1536 blocks
    conv1x1_wmma<<<g1, 128, 0, stream>>>(feat1, Wq, bq, qws);
    conv1x1_wmma<<<g1, 128, 0, stream>>>(feat2, Wk, bk, kws);

    dim3 g2((BD * HWD) / 256);                            // 384 blocks
    corr_softmax<<<g2, 256, 0, stream>>>(qws, kws, out);
}
